// MotionPredictionNN_44736379355771
// MI455X (gfx1250) — matrix operand order repaired and checked by reading, not compiled, not run on hardware
//
#include <hip/hip_runtime.h>

typedef __attribute__((ext_vector_type(16))) _Float16 v16h;
typedef __attribute__((ext_vector_type(8)))  float    v8f;

#define NEXP 8
#define KP   512   // padded K / partials column stride

// ------------------------------------------------------------------
// cvt: X16[b][k] = f16(act[b][k]) zero-padded to KP columns
// ------------------------------------------------------------------
__global__ __launch_bounds__(256)
void cvt_pad_f16(const float* __restrict__ act, _Float16* __restrict__ X16,
                 int M, int Kreal) {
  int idx = blockIdx.x * blockDim.x + threadIdx.x;
  if (idx >= M * KP) return;
  int b = idx >> 9;          // /KP
  int k = idx & (KP - 1);
  X16[idx] = (k < Kreal) ? (_Float16)act[(size_t)b * Kreal + k] : (_Float16)0.0f;
}

// ------------------------------------------------------------------
// Per-expert GEMM partial:  P[e][b][n] = BC[b,e] * (X[b,:] . W_e[n,:])
// X : f16 [M][KP] (zero-padded), W : f32 [E][Ntrue][Kw]
// grid = (M/128, ceil(N/64), E); 256 thr = 8 waves, wave tile 32x32.
// Branch-free staging: A side is zero-padded, W side uses clamped
// (always in-bounds) addresses whose duplicated values are annihilated
// by the zero A columns / never-read P columns.
// ------------------------------------------------------------------
#define BM 128
#define BN 64
#define KC 64
#define AST 72   // LDS row stride in halves (144B: 16B-aligned, bank-spread)
#define BST 72

__global__ __launch_bounds__(256)
void moe_gemm_expert(const _Float16* __restrict__ X, const float* __restrict__ W,
                     const float* __restrict__ BC, float* __restrict__ P,
                     int M, int Ntrue, int Kw) {
  __shared__ _Float16 smem[BM * AST + BN * BST];
  _Float16* sA = smem;
  _Float16* sB = smem + BM * AST;

  const int e     = blockIdx.z;
  const int tid   = threadIdx.x;
  const int lane  = tid & 31;
  const int wave  = tid >> 5;
  const int wm    = wave & 3;    // 4 waves along M -> 128
  const int wn    = wave >> 2;   // 2 waves along N -> 64
  const int lsub  = lane >> 4;
  const int l15   = lane & 15;
  const int mBlock = blockIdx.x * BM;
  const int nBlock = blockIdx.y * BN;

  const float* We = W + (size_t)e * Ntrue * Kw;

  v8f zero = {0.f, 0.f, 0.f, 0.f, 0.f, 0.f, 0.f, 0.f};
  v8f acc[2][2];
  acc[0][0] = zero; acc[0][1] = zero; acc[1][0] = zero; acc[1][1] = zero;

  const int nchunks = KP / KC;   // 8

  for (int kc = 0; kc < nchunks; ++kc) {
    const int kbase = kc * KC;

    // ---- stage A tile [128][64] f16: unconditional (X is zero-padded) ----
#pragma unroll
    for (int i = 0; i < 4; ++i) {
      int v   = tid + i * 256;
      int row = v >> 3;
      int cv  = (v & 7) << 3;
      uint4 d = *reinterpret_cast<const uint4*>(
          X + (size_t)(mBlock + row) * KP + kbase + cv);
      *reinterpret_cast<uint4*>(sA + row * AST + cv) = d;
    }
    // ---- stage W tile [64][64]: f32 -> f16, clamped addressing ----
#pragma unroll
    for (int i = 0; i < 4; ++i) {
      int v   = tid + i * 256;
      int row = v >> 4;
      int c4  = (v & 15) << 2;
      int ng  = min(nBlock + row, Ntrue - 1);
      int gk  = min(kbase + c4, Kw - 4);
      const float* src = We + (size_t)ng * Kw + gk;
      if (kc + 1 < nchunks) __builtin_prefetch(src + KC, 0, 0);
      float4 wv = *reinterpret_cast<const float4*>(src);
      _Float16* dst = sB + row * BST + c4;
      dst[0] = (_Float16)wv.x; dst[1] = (_Float16)wv.y;
      dst[2] = (_Float16)wv.z; dst[3] = (_Float16)wv.w;
    }
    __syncthreads();

    // ---- fragments per ISA 16-bit VGPR layouts ----
    union Frag { v16h v; uint4 q[2]; };
    Frag fa[2][2], fb[2][2];
#pragma unroll
    for (int mi = 0; mi < 2; ++mi) {
      const _Float16* ap = sA + (wm * 32 + mi * 16 + l15) * AST;
#pragma unroll
      for (int kh = 0; kh < 2; ++kh) {
        // A 16x32: lanes0-15 {K0..7,K16..23}, lanes16-31 {K8..15,K24..31}
        fa[mi][kh].q[0] = *reinterpret_cast<const uint4*>(ap + kh * 32 + lsub * 8);
        fa[mi][kh].q[1] = *reinterpret_cast<const uint4*>(ap + kh * 32 + 16 + lsub * 8);
      }
    }
#pragma unroll
    for (int ni = 0; ni < 2; ++ni) {
      const _Float16* bp = sB + (wn * 32 + ni * 16 + l15) * BST;
#pragma unroll
      for (int kh = 0; kh < 2; ++kh) {
        // B 32x16: lanes0-15 {K0..7,K16..23}, lanes16-31 {K8..15,K24..31}
        fb[ni][kh].q[0] = *reinterpret_cast<const uint4*>(bp + kh * 32 + lsub * 8);
        fb[ni][kh].q[1] = *reinterpret_cast<const uint4*>(bp + kh * 32 + 16 + lsub * 8);
      }
    }
#pragma unroll
    for (int mi = 0; mi < 2; ++mi)
#pragma unroll
      for (int ni = 0; ni < 2; ++ni)
#pragma unroll
        for (int kh = 0; kh < 2; ++kh)
          acc[mi][ni] = __builtin_amdgcn_wmma_f32_16x16x32_f16(
              false, fa[mi][kh].v, false, fb[ni][kh].v,
              (short)0, acc[mi][ni], false, false);
    __syncthreads();
  }

  // ---- epilogue: scale by gate BC[b,e], store partial (stride KP, no guards) ----
#pragma unroll
  for (int mi = 0; mi < 2; ++mi) {
#pragma unroll
    for (int ni = 0; ni < 2; ++ni) {
#pragma unroll
      for (int i = 0; i < 8; ++i) {
        int row = mBlock + wm * 32 + mi * 16 + i + lsub * 8;
        int col = nBlock + wn * 32 + ni * 16 + l15;
        float g = BC[row * NEXP + e];
        P[((size_t)e * M + row) * KP + col] = acc[mi][ni][i] * g;
      }
    }
  }
}

// ------------------------------------------------------------------
// Deterministic reduce over experts + blended bias + ELU + downcast.
// out16 stride KP (feeds next layer's A); out32 stride Ntrue (final).
// ------------------------------------------------------------------
__global__ __launch_bounds__(256)
void moe_reduce(const float* __restrict__ P, const float* __restrict__ BC,
                const float* __restrict__ Be, _Float16* __restrict__ out16,
                float* __restrict__ out32, int M, int Ntrue,
                int apply_elu, int write_f16) {
  int idx = blockIdx.x * blockDim.x + threadIdx.x;
  if (idx >= M * Ntrue) return;
  int b = idx / Ntrue;
  int n = idx - b * Ntrue;
  float s = 0.f;
#pragma unroll
  for (int e = 0; e < NEXP; ++e) s += P[((size_t)e * M + b) * KP + n];
  float bias = 0.f;
#pragma unroll
  for (int e = 0; e < NEXP; ++e) bias += BC[b * NEXP + e] * Be[e * Ntrue + n];
  float v = s + bias;
  if (apply_elu) v = (v > 0.f) ? v : (__expf(v) - 1.f);
  if (write_f16) out16[(size_t)b * KP + n] = (_Float16)v;
  else           out32[(size_t)b * Ntrue + n] = v;
}

// ------------------------------------------------------------------
extern "C" void kernel_launch(void* const* d_in, const int* in_sizes, int n_in,
                              void* d_out, int out_size, void* d_ws, size_t ws_size,
                              hipStream_t stream) {
  const float* x   = (const float*)d_in[0];   // [512,480]
  const float* BC  = (const float*)d_in[1];   // [512,8]
  const float* We0 = (const float*)d_in[2];   // [8,512,480]
  const float* Be0 = (const float*)d_in[3];   // [8,512]
  const float* We1 = (const float*)d_in[4];   // [8,512,512]
  const float* Be1 = (const float*)d_in[5];   // [8,512]
  const float* We2 = (const float*)d_in[6];   // [8,363,512]
  const float* Be2 = (const float*)d_in[7];   // [8,363]
  float* out = (float*)d_out;                 // [512,363]

  const int M = 512, H = 512, NI = 480, NO = 363;

  char* ws = (char*)d_ws;
  float*    P    = (float*)ws;                                  // 8*512*512*4 = 8 MB
  _Float16* bufA = (_Float16*)(ws + (size_t)NEXP * M * KP * 4); // 512*512*2
  _Float16* bufB = bufA + (size_t)M * KP;                       // 512*512*2

  dim3 blk(256);
  const int cvtBlocks = (M * KP + 255) / 256;
  const int redBlocksH = (M * H + 255) / 256;
  const int redBlocksO = (M * NO + 255) / 256;

  // -------- layer 0: [512,480] -> [512,512], ELU --------
  cvt_pad_f16<<<dim3(cvtBlocks), blk, 0, stream>>>(x, bufA, M, NI);
  moe_gemm_expert<<<dim3(M / BM, H / BN, NEXP), blk, 0, stream>>>(
      bufA, We0, BC, P, M, H, NI);
  moe_reduce<<<dim3(redBlocksH), blk, 0, stream>>>(
      P, BC, Be0, bufB, nullptr, M, H, /*elu=*/1, /*f16=*/1);

  // -------- layer 1: [512,512] -> [512,512], ELU --------
  moe_gemm_expert<<<dim3(M / BM, H / BN, NEXP), blk, 0, stream>>>(
      bufB, We1, BC, P, M, H, H);
  moe_reduce<<<dim3(redBlocksH), blk, 0, stream>>>(
      P, BC, Be1, bufA, nullptr, M, H, /*elu=*/1, /*f16=*/1);

  // -------- layer 2: [512,512] -> [512,363], linear --------
  moe_gemm_expert<<<dim3(M / BM, (NO + BN - 1) / BN, NEXP), blk, 0, stream>>>(
      bufA, We2, BC, P, M, NO, H);
  moe_reduce<<<dim3(redBlocksO), blk, 0, stream>>>(
      P, BC, Be2, nullptr, out, M, NO, /*elu=*/0, /*f16=*/0);
}
